// AdvancedGenuineTransformer_35485019799944
// MI455X (gfx1250) — matrix operand order repaired and checked by reading, not compiled, not run on hardware
//
#include <hip/hip_runtime.h>
#include <hip/hip_bf16.h>

// ---------------------------------------------------------------------------
// MI455X (gfx1250) transformer forward.
// Dense math -> v_wmma_f32_16x16x32_bf16 (wave32 WMMA).
// Tile staging -> global_load_async_to_lds_b128 (ASYNCcnt DMA).
// Weights pre-transposed+converted to bf16 [N][K] once per launch.
// ---------------------------------------------------------------------------

typedef __attribute__((ext_vector_type(16))) __bf16 v16bf;
typedef __attribute__((ext_vector_type(8)))  __bf16 v8bf;
typedef __attribute__((ext_vector_type(8)))  float  v8f;

union FragAB { v16bf v; v8bf h[2]; };

constexpr int kB  = 16;
constexpr int kS  = 128;
constexpr int kD  = 1024;
constexpr int kH  = 16;
constexpr int kV  = 32000;
constexpr int kHD = 64;            // head dim
constexpr int kM  = kB * kS;       // 2048 token rows
constexpr int kFF = 4 * kD;        // 4096

// GEMM tiling: 128x128 block tile, K-step 32 (= WMMA K), 8 waves (4x2).
constexpr int TBM  = 128;
constexpr int TBN  = 128;
constexpr int TBK  = 32;
constexpr int TLDA = 40;           // padded LDS strides (conflict-free b128 reads)
constexpr int TLDB = 40;

// ---------------------------------------------------------------------------
// CDNA5 async global->LDS copy (guarded: falls back to register staging)
// Signature (from hipcc diagnostic): first param is v4i in __device__ (AS1),
// so: (v4i AS1* gsrc, v4i AS3* ldst, imm offset, imm cpol)
// ---------------------------------------------------------------------------
#if defined(__has_builtin)
#if __has_builtin(__builtin_amdgcn_global_load_async_to_lds_b128)
#define GFX1250_ASYNC_LDS 1
#endif
#endif

#ifdef GFX1250_ASYNC_LDS
typedef int v4i __attribute__((vector_size(16)));
typedef __attribute__((address_space(1))) v4i* gv4i_p;
typedef __attribute__((address_space(3))) v4i* lv4i_p;

template <int OFF>
static __device__ __forceinline__ void async_cp16(const __bf16* g, __bf16* l) {
  // LDS[l + OFF] = MEM[g + OFF], 16 bytes per lane, tracked by ASYNCcnt
  __builtin_amdgcn_global_load_async_to_lds_b128(
      (gv4i_p)(void*)g, (lv4i_p)(void*)l, OFF, 0);
}
#endif

static __device__ __forceinline__ void async_wait0() {
#ifdef GFX1250_ASYNC_LDS
#if __has_builtin(__builtin_amdgcn_s_wait_asynccnt)
  __builtin_amdgcn_s_wait_asynccnt(0);
#else
  asm volatile("s_wait_asynccnt 0" ::: "memory");
#endif
#endif
}

// ---------------------------------------------------------------------------
// Embedding gather: h[row,:] = emb[x[row],:]
// ---------------------------------------------------------------------------
__global__ void embed_kernel(const int* __restrict__ x,
                             const float* __restrict__ emb,
                             float* __restrict__ h) {
  const int row = blockIdx.x;
  const long tok = (long)x[row];
  const float* src = emb + tok * (long)kD;
  float* dst = h + (long)row * kD;
  for (int d = threadIdx.x; d < kD; d += blockDim.x) dst[d] = src[d];
}

// ---------------------------------------------------------------------------
// One-time weight prep: W_f32[K][N] -> Wt_bf16[N][K] (32x32 LDS tiles)
// ---------------------------------------------------------------------------
__global__ __launch_bounds__(256) void transpose_bf16_kernel(
    const float* __restrict__ W, __bf16* __restrict__ Wt, int K, int N) {
  __shared__ float tile[32][33];
  const int kb = blockIdx.y * 32;
  const int nb = blockIdx.x * 32;
  const int tx = threadIdx.x & 31;
  const int ty = threadIdx.x >> 5;  // 0..7
  for (int r = ty; r < 32; r += 8)
    tile[r][tx] = W[(long)(kb + r) * N + nb + tx];
  __syncthreads();
  for (int r = ty; r < 32; r += 8)
    Wt[(long)(nb + r) * K + kb + tx] = (__bf16)tile[tx][r];
}

// ---------------------------------------------------------------------------
// LayerNorm (per row) with fused fp32 -> bf16 output (WMMA A operand).
// ---------------------------------------------------------------------------
__global__ __launch_bounds__(256) void ln_bf16_kernel(
    const float* __restrict__ x, const float* __restrict__ g,
    const float* __restrict__ b, __bf16* __restrict__ out, int D) {
  __shared__ float sh1[256];
  __shared__ float sh2[256];
  const int row = blockIdx.x;
  const float* xr = x + (long)row * D;
  float s = 0.f, s2 = 0.f;
  for (int d = threadIdx.x; d < D; d += 256) {
    float t = xr[d];
    s += t; s2 += t * t;
  }
  sh1[threadIdx.x] = s; sh2[threadIdx.x] = s2;
  __syncthreads();
  for (int off = 128; off > 0; off >>= 1) {
    if (threadIdx.x < off) {
      sh1[threadIdx.x] += sh1[threadIdx.x + off];
      sh2[threadIdx.x] += sh2[threadIdx.x + off];
    }
    __syncthreads();
  }
  const float mean = sh1[0] / (float)D;
  const float var  = sh2[0] / (float)D - mean * mean;
  const float inv  = rsqrtf(var + 1e-5f);
  __bf16* orow = out + (long)row * D;
  for (int d = threadIdx.x; d < D; d += 256)
    orow[d] = (__bf16)((xr[d] - mean) * inv * g[d] + b[d]);
}

// ---------------------------------------------------------------------------
// fp32 -> bf16 elementwise (prep next GEMM's A operand)
// ---------------------------------------------------------------------------
__global__ void f32_to_bf16_kernel(const float* __restrict__ src,
                                   __bf16* __restrict__ dst, long n) {
  long i = (long)blockIdx.x * blockDim.x + threadIdx.x;
  if (i < n) dst[i] = (__bf16)src[i];
}

// ---------------------------------------------------------------------------
// WMMA GEMM: C[M,N] = act(A_bf16[M,K] @ Bt_bf16[N][K]^T + bias) + resid
//   - A row-major [M][K], Bt pre-transposed [N][K]; symmetric 32B staging
//   - 128x128 block, double-buffered LDS, async DMA staging when available
//   - 8 waves (4x2), 2x4 WMMA accumulators per wave
//   - epilogue fuses bias / exact GELU / residual
// ---------------------------------------------------------------------------
__global__ __launch_bounds__(256) void gemm_bf16_wmma(
    const __bf16* __restrict__ A, const __bf16* __restrict__ Bt,
    const float* __restrict__ bias, const float* __restrict__ resid,
    float* __restrict__ C, int M, int N, int K, int act) {
  __shared__ __align__(16) __bf16 As[2][TBM * TLDA];
  __shared__ __align__(16) __bf16 Bs[2][TBN * TLDB];

  const int tid  = threadIdx.x;
  const int lane = tid & 31;
  const int wid  = tid >> 5;       // 0..7
  const int wm   = wid & 3;        // 4 waves along M (32 rows each)
  const int wn   = wid >> 2;       // 2 waves along N (64 cols each)
  const int m0   = blockIdx.y * TBM;
  const int n0   = blockIdx.x * TBN;

  // staging: thread copies one 32B A segment + one 32B B segment per tile
  const int arow = tid >> 1;             // 0..127 (also the B column)
  const int aseg = (tid & 1) << 4;       // 0 / 16 elements

  const __bf16* gaBase = A  + (long)(m0 + arow) * K + aseg;
  const __bf16* gbBase = Bt + (long)(n0 + arow) * K + aseg;
  __bf16* lA0 = &As[0][arow * TLDA + aseg];
  __bf16* lB0 = &Bs[0][arow * TLDB + aseg];
  __bf16* lA1 = &As[1][arow * TLDA + aseg];
  __bf16* lB1 = &Bs[1][arow * TLDB + aseg];

  v8f zacc = {};
  v8f acc[2][4];
#pragma unroll
  for (int i = 0; i < 2; ++i)
#pragma unroll
    for (int j = 0; j < 4; ++j) acc[i][j] = zacc;

  const int KT = K / TBK;

  auto stage = [&](int kt, int buf) {
    const __bf16* ga = gaBase + (long)kt * TBK;
    const __bf16* gb = gbBase + (long)kt * TBK;
    __bf16* la = buf ? lA1 : lA0;
    __bf16* lb = buf ? lB1 : lB0;
#ifdef GFX1250_ASYNC_LDS
    async_cp16<0>(ga, la);
    async_cp16<16>(ga, la);
    async_cp16<0>(gb, lb);
    async_cp16<16>(gb, lb);
#else
    v8bf a0 = *(const v8bf*)ga;
    v8bf a1 = *(const v8bf*)(ga + 8);
    v8bf b0 = *(const v8bf*)gb;
    v8bf b1 = *(const v8bf*)(gb + 8);
    *(v8bf*)la       = a0;
    *(v8bf*)(la + 8) = a1;
    *(v8bf*)lb       = b0;
    *(v8bf*)(lb + 8) = b1;
#endif
  };

  stage(0, 0);
  async_wait0();
  __syncthreads();

  const int kh = lane >> 4;   // K-half select (contiguous k order, same for A and B)
  const int ml = lane & 15;

  for (int kt = 0; kt < KT; ++kt) {
    const int cur = kt & 1;
    const bool more = (kt + 1) < KT;
    if (more) {  // DMA next tile while WMMAs below run on `cur`
      stage(kt + 1, cur ^ 1);
      if (kt + 2 < KT) {  // global_prefetch_b8 two tiles ahead into L2
        __builtin_prefetch(gbBase + (long)(kt + 2) * TBK, 0, 1);
        __builtin_prefetch(gaBase + (long)(kt + 2) * TBK, 0, 1);
      }
    }

    // ---- WMMA compute on buffer `cur` ----
    {
      FragAB a0, a1, b0, b1, b2, b3;
      // A fragment (16x32 bf16): lane row = ml, 16 contiguous K per lane (half kh)
      const __bf16* pa = &As[cur][(wm * 32 + ml) * TLDA + kh * 16];
      a0.h[0] = *(const v8bf*)pa;
      a0.h[1] = *(const v8bf*)(pa + 8);
      const __bf16* pa2 = pa + 16 * TLDA;
      a1.h[0] = *(const v8bf*)pa2;
      a1.h[1] = *(const v8bf*)(pa2 + 8);
      // B fragment (32x16 bf16): lane col = ml, 16 contiguous K per lane
      const __bf16* pb = &Bs[cur][(wn * 64 + ml) * TLDB + kh * 16];
      b0.h[0] = *(const v8bf*)pb;
      b0.h[1] = *(const v8bf*)(pb + 8);
      b1.h[0] = *(const v8bf*)(pb + 16 * TLDB);
      b1.h[1] = *(const v8bf*)(pb + 16 * TLDB + 8);
      b2.h[0] = *(const v8bf*)(pb + 32 * TLDB);
      b2.h[1] = *(const v8bf*)(pb + 32 * TLDB + 8);
      b3.h[0] = *(const v8bf*)(pb + 48 * TLDB);
      b3.h[1] = *(const v8bf*)(pb + 48 * TLDB + 8);

      acc[0][0] = __builtin_amdgcn_wmma_f32_16x16x32_bf16(false, a0.v, false, b0.v, (short)0, acc[0][0], false, false);
      acc[0][1] = __builtin_amdgcn_wmma_f32_16x16x32_bf16(false, a0.v, false, b1.v, (short)0, acc[0][1], false, false);
      acc[0][2] = __builtin_amdgcn_wmma_f32_16x16x32_bf16(false, a0.v, false, b2.v, (short)0, acc[0][2], false, false);
      acc[0][3] = __builtin_amdgcn_wmma_f32_16x16x32_bf16(false, a0.v, false, b3.v, (short)0, acc[0][3], false, false);
      acc[1][0] = __builtin_amdgcn_wmma_f32_16x16x32_bf16(false, a1.v, false, b0.v, (short)0, acc[1][0], false, false);
      acc[1][1] = __builtin_amdgcn_wmma_f32_16x16x32_bf16(false, a1.v, false, b1.v, (short)0, acc[1][1], false, false);
      acc[1][2] = __builtin_amdgcn_wmma_f32_16x16x32_bf16(false, a1.v, false, b2.v, (short)0, acc[1][2], false, false);
      acc[1][3] = __builtin_amdgcn_wmma_f32_16x16x32_bf16(false, a1.v, false, b3.v, (short)0, acc[1][3], false, false);
    }

    if (more) async_wait0();  // own wave's DMA done; barrier covers the rest
    __syncthreads();
  }

  // ---- epilogue: C/D layout -> (row, col); fuse bias / gelu / residual ----
  const int mh = kh * 8;
#pragma unroll
  for (int mt = 0; mt < 2; ++mt) {
#pragma unroll
    for (int nt = 0; nt < 4; ++nt) {
#pragma unroll
      for (int j = 0; j < 8; ++j) {
        const int row = m0 + wm * 32 + mt * 16 + mh + j;
        const int col = n0 + wn * 64 + nt * 16 + ml;
        float val = acc[mt][nt][j];
        if (bias)  val += bias[col];
        if (act)   val = 0.5f * val * (1.0f + erff(val * 0.70710678118f));
        if (resid) val += resid[(long)row * N + col];
        C[(long)row * N + col] = val;
      }
    }
  }
}

// ---------------------------------------------------------------------------
// Attention (per b,h block): fused RoPE, streaming softmax, entropy, o = w@v
// ---------------------------------------------------------------------------
__global__ __launch_bounds__(128) void attn_kernel(
    const float* __restrict__ q, const float* __restrict__ k,
    const float* __restrict__ v, float* __restrict__ o,
    float* __restrict__ ent) {
  __shared__ float  ks[kS][kHD + 2];
  __shared__ __bf16 vs[kS][kHD + 2];

  const int b  = blockIdx.x / kH;
  const int h  = blockIdx.x % kH;
  const int m0 = b * kS;
  const int c0 = h * kHD;
  const int i  = threadIdx.x;

  float qv[kHD];
  {
    const float* kr = k + (long)(m0 + i) * kD + c0;
    const float* vr = v + (long)(m0 + i) * kD + c0;
    const float* qr = q + (long)(m0 + i) * kD + c0;
    const float pos = (float)i;
#pragma unroll
    for (int d2 = 0; d2 < kHD / 2; ++d2) {
      const float inv = __expf(-(2.0f * d2 / (float)kHD) * 9.210340371976184f);
      const float ang = pos * inv;
      float sn, cs;
      __sincosf(ang, &sn, &cs);
      const float xr = kr[2 * d2], xi = kr[2 * d2 + 1];
      ks[i][2 * d2]     = xr * cs - xi * sn;
      ks[i][2 * d2 + 1] = xr * sn + xi * cs;
      vs[i][2 * d2]     = (__bf16)vr[2 * d2];
      vs[i][2 * d2 + 1] = (__bf16)vr[2 * d2 + 1];
      const float qxr = qr[2 * d2], qxi = qr[2 * d2 + 1];
      qv[2 * d2]     = qxr * cs - qxi * sn;
      qv[2 * d2 + 1] = qxr * sn + qxi * cs;
    }
  }
  __syncthreads();

  const float scale = 0.125f;  // 1/sqrt(64)
  float mx = -3.0e38f;
  for (int j = 0; j < kS; ++j) {
    float dot = 0.f;
#pragma unroll
    for (int d = 0; d < kHD; ++d) dot += qv[d] * ks[j][d];
    mx = fmaxf(mx, dot * scale);
  }
  float t1 = 0.f, t2 = 0.f;
  float oa[kHD];
#pragma unroll
  for (int d = 0; d < kHD; ++d) oa[d] = 0.f;
  for (int j = 0; j < kS; ++j) {
    float dot = 0.f;
#pragma unroll
    for (int d = 0; d < kHD; ++d) dot += qv[d] * ks[j][d];
    const float sc = dot * scale - mx;
    const float e  = __expf(sc);
    t1 += e;
    t2 += e * sc;
#pragma unroll
    for (int d = 0; d < kHD; ++d) oa[d] += e * (float)vs[j][d];
  }
  const float inv1 = 1.0f / t1;
  float* orow = o + (long)(m0 + i) * kD + c0;
#pragma unroll
  for (int d = 0; d < kHD; ++d) orow[d] = oa[d] * inv1;
  ent[(long)(b * kS + i) * kH + h] = (__logf(t1) - t2 * inv1) * 1.4426950408889634f;
}

// ---------------------------------------------------------------------------
// Entropy gate: mean over (b,s) of var over heads (ddof=1), summed per layer
// ---------------------------------------------------------------------------
__global__ void entvar_kernel(const float* __restrict__ ent,
                              float* __restrict__ gsum) {
  const int r = blockIdx.x * blockDim.x + threadIdx.x;
  if (r >= kM) return;
  float s = 0.f, s2 = 0.f;
  for (int j = 0; j < kH; ++j) {
    const float e = ent[(long)r * kH + j];
    s += e; s2 += e * e;
  }
  const float mean = s / (float)kH;
  const float var  = (s2 - (float)kH * mean * mean) / (float)(kH - 1);
  atomicAdd(gsum, var * (1.0f / (float)kM));
}

__global__ void zero_kernel(float* __restrict__ p, int n) {
  const int i = blockIdx.x * blockDim.x + threadIdx.x;
  if (i < n) p[i] = 0.f;
}

// h = (g >= 0.6) ? pass1 : pass2   (device-side lax.cond resolution)
__global__ void select_kernel(const float* __restrict__ hsave,
                              float* __restrict__ hcur,
                              const float* __restrict__ gsum) {
  const float g = gsum[0] * (1.0f / 3.0f);
  const long i = (long)blockIdx.x * blockDim.x + threadIdx.x;
  if (i < (long)kM * kD) hcur[i] = (g >= 0.6f) ? hsave[i] : hcur[i];
}

// ---------------------------------------------------------------------------
// Host orchestration
// ---------------------------------------------------------------------------
extern "C" void kernel_launch(void* const* d_in, const int* in_sizes, int n_in,
                              void* d_out, int out_size, void* d_ws, size_t ws_size,
                              hipStream_t stream) {
  (void)in_sizes; (void)n_in; (void)out_size; (void)ws_size;

  const int*   x    = (const int*)  d_in[0];
  const float* emb  = (const float*)d_in[1];
  const float* wq   = (const float*)d_in[2];
  const float* wk   = (const float*)d_in[3];
  const float* wv   = (const float*)d_in[4];
  const float* wo   = (const float*)d_in[5];
  const float* ln1g = (const float*)d_in[6];
  const float* ln1b = (const float*)d_in[7];
  const float* w1   = (const float*)d_in[8];
  const float* b1   = (const float*)d_in[9];
  const float* w2   = (const float*)d_in[10];
  const float* b2   = (const float*)d_in[11];
  const float* ln2g = (const float*)d_in[12];
  const float* ln2b = (const float*)d_in[13];
  const float* fow  = (const float*)d_in[14];
  const float* fob  = (const float*)d_in[15];
  float* out = (float*)d_out;

  const long dd = (long)kD * kD;

  // workspace carve-up (all 16B aligned)
  char* wsp = (char*)d_ws;
  float* h     = (float*)wsp; wsp += sizeof(float) * (long)kM * kD;
  float* hsave = (float*)wsp; wsp += sizeof(float) * (long)kM * kD;
  float* qb    = (float*)wsp; wsp += sizeof(float) * (long)kM * kD;
  float* kb    = (float*)wsp; wsp += sizeof(float) * (long)kM * kD;
  float* vb    = (float*)wsp; wsp += sizeof(float) * (long)kM * kD;
  float* ob    = (float*)wsp; wsp += sizeof(float) * (long)kM * kD;
  float* mid   = (float*)wsp; wsp += sizeof(float) * (long)kM * kFF;
  float* ent   = (float*)wsp; wsp += sizeof(float) * (long)kM * kH;
  float* gsum  = (float*)wsp; wsp += sizeof(float) * 16;
  __bf16* zb   = (__bf16*)wsp; wsp += sizeof(__bf16) * (long)kM * kFF;
  // pre-transposed bf16 weights [N][K]
  __bf16* wtq  = (__bf16*)wsp; wsp += sizeof(__bf16) * 6 * dd;
  __bf16* wtk  = (__bf16*)wsp; wsp += sizeof(__bf16) * 6 * dd;
  __bf16* wtv  = (__bf16*)wsp; wsp += sizeof(__bf16) * 6 * dd;
  __bf16* wto  = (__bf16*)wsp; wsp += sizeof(__bf16) * 6 * dd;
  __bf16* wt1  = (__bf16*)wsp; wsp += sizeof(__bf16) * 6 * (long)kD * kFF;
  __bf16* wt2  = (__bf16*)wsp; wsp += sizeof(__bf16) * 6 * (long)kD * kFF;
  __bf16* wtf  = (__bf16*)wsp; wsp += sizeof(__bf16) * (long)kD * kV;

  auto xpose = [&](const float* W, __bf16* Wt, int K, int N) {
    transpose_bf16_kernel<<<dim3(N / 32, K / 32), 256, 0, stream>>>(W, Wt, K, N);
  };
  for (int i = 0; i < 6; ++i) {
    xpose(wq + i * dd, wtq + i * dd, kD, kD);
    xpose(wk + i * dd, wtk + i * dd, kD, kD);
    xpose(wv + i * dd, wtv + i * dd, kD, kD);
    xpose(wo + i * dd, wto + i * dd, kD, kD);
    xpose(w1 + i * (long)kD * kFF, wt1 + i * (long)kD * kFF, kD, kFF);
    xpose(w2 + i * (long)kD * kFF, wt2 + i * (long)kD * kFF, kFF, kD);
  }
  xpose(fow, wtf, kD, kV);

  auto gemm = [&](const __bf16* A, const __bf16* Bt, const float* bias,
                  const float* res, float* C, int N, int K, int act) {
    dim3 grid(N / TBN, kM / TBM);
    gemm_bf16_wmma<<<grid, 256, 0, stream>>>(A, Bt, bias, res, C, kM, N, K, act);
  };

  auto layer = [&](int i, bool with_ent) {
    ln_bf16_kernel<<<kM, 256, 0, stream>>>(h, ln1g + (long)i * kD, ln1b + (long)i * kD, zb, kD);
    gemm(zb, wtq + i * dd, nullptr, nullptr, qb, kD, kD, 0);
    gemm(zb, wtk + i * dd, nullptr, nullptr, kb, kD, kD, 0);
    gemm(zb, wtv + i * dd, nullptr, nullptr, vb, kD, kD, 0);
    attn_kernel<<<kB * kH, 128, 0, stream>>>(qb, kb, vb, ob, ent);
    f32_to_bf16_kernel<<<(kM * kD) / 256, 256, 0, stream>>>(ob, zb, (long)kM * kD);
    gemm(zb, wto + i * dd, nullptr, h, h, kD, kD, 0);          // + residual
    if (with_ent)
      entvar_kernel<<<kM / 256, 256, 0, stream>>>(ent, gsum);
    ln_bf16_kernel<<<kM, 256, 0, stream>>>(h, ln2g + (long)i * kD, ln2b + (long)i * kD, zb, kD);
    gemm(zb, wt1 + i * (long)kD * kFF, b1 + (long)i * kFF, nullptr, mid, kFF, kD, 1); // gelu
    f32_to_bf16_kernel<<<(kM * kFF) / 256, 256, 0, stream>>>(mid, zb, (long)kM * kFF);
    gemm(zb, wt2 + i * (long)kD * kFF, b2 + (long)i * kD, h, h, kD, kFF, 0); // + residual
  };

  embed_kernel<<<kM, 256, 0, stream>>>(x, emb, h);
  zero_kernel<<<1, 32, 0, stream>>>(gsum, 16);

  // reasoning pass 1 (accumulates gate g into gsum)
  for (int i = 0; i < 3; ++i) layer(i, true);
  (void)hipMemcpyAsync(hsave, h, sizeof(float) * (long)kM * kD,
                       hipMemcpyDeviceToDevice, stream);
  // reasoning pass 2 (speculative; selected only if g < 0.6)
  for (int i = 0; i < 3; ++i) layer(i, false);
  select_kernel<<<(kM * kD) / 256, 256, 0, stream>>>(hsave, h, gsum);

  // layers 3..5
  for (int i = 3; i < 6; ++i) layer(i, false);

  // final vocab projection: out = h @ fo_w + fo_b   (2048 x 32000)
  f32_to_bf16_kernel<<<(kM * kD) / 256, 256, 0, stream>>>(h, zb, (long)kM * kD);
  gemm(zb, wtf, fob, nullptr, out, kV, kD, 0);
}
